// DirectionalMessagePassing_27041114095748
// MI455X (gfx1250) — matrix operand order repaired and checked by reading, not compiled, not run on hardware
//
#include <hip/hip_runtime.h>
#include <stdint.h>

#define NN 100000
#define NE 640000
#define H  128
#define RAD 50
#define CUT 10.0f

// LDS row pitches in bf16 elements (dword pitch % 64 == 4 or 36 -> the 16
// A-rows start on distinct banks).
#define FP_E 328   // edge feature tile: 320 used, pitch 328 (164 dwords)
#define FP_N 264   // node feature tile: 256 used, pitch 264 (132 dwords)
#define HP   136   // hidden tiles: 128 used, pitch 136 (68 dwords)

typedef __attribute__((ext_vector_type(16))) __bf16 v16bf;
typedef __attribute__((ext_vector_type(8)))  float  v8f;

__device__ __forceinline__ float silu_f(float v) {
  return v * __frcp_rn(1.0f + __expf(-v));
}

__device__ __forceinline__ uint32_t pk_bf16(float a, float b) {
  union { __bf16 h[2]; uint32_t u; } cv;
  cv.h[0] = (__bf16)a;
  cv.h[1] = (__bf16)b;
  return cv.u;
}

// A fragment from a bf16 LDS row already stored in K-order: two 16-byte runs,
// K = kt*32 + half*8 + [0..7]  and  K = kt*32 + 16 + half*8 + [0..7].
// (CDNA5 ISA 7.12.2: 16-bit A 16x32 layout.)  No cvts: raw ds_load_b128 x2.
__device__ __forceinline__ v16bf load_a_bf(const __bf16* rowp, int kt, int half) {
  union { v16bf v; uint4 u[2]; } t;
  t.u[0] = *(const uint4*)(rowp + kt*32 + half*8);
  t.u[1] = *(const uint4*)(rowp + kt*32 + 16 + half*8);
  return t.v;
}

// Half-N dual-M GEMM pass: 4 N-tiles (nh selects which half), two 16-row
// M-tiles.  Each B fragment (1KB/wave from L2) feeds two WMMAs.
// pw packed kt-major: fragment(kt,nt) at v16bf index (kt*8+nt)*32 + lane.
// kt loop only partially unrolled to bound the live register set (< 256 VGPRs,
// avoiding s_set_vgpr_msb mode switching).
template<int KT>
__device__ __forceinline__ void gemm_half(const __bf16* tile, int pitch,
                                          const uint32_t* __restrict__ pw,
                                          const float* __restrict__ bias,
                                          int nh, int lane,
                                          v8f c0[4], v8f c1[4]) {
  const int nl = lane & 15, half = lane >> 4;
  const __bf16* r0 = tile + (size_t)nl * pitch;
  const __bf16* r1 = tile + (size_t)(16 + nl) * pitch;
#pragma unroll
  for (int q = 0; q < 4; ++q) {
    const float bv = bias[(nh*4 + q)*16 + nl];
    v8f b = {bv, bv, bv, bv, bv, bv, bv, bv};
    c0[q] = b;
    c1[q] = b;
  }
#pragma unroll 2
  for (int kt = 0; kt < KT; ++kt) {
    v16bf a0 = load_a_bf(r0, kt, half);
    v16bf a1 = load_a_bf(r1, kt, half);
    const v16bf* bp = (const v16bf*)pw + (size_t)(kt*8 + nh*4) * 32 + lane;
#pragma unroll
    for (int q = 0; q < 4; ++q) {
      v16bf b = bp[q * 32];
      c0[q] = __builtin_amdgcn_wmma_f32_16x16x32_bf16(
                  false, a0, false, b, (short)0, c0[q], false, false);
      c1[q] = __builtin_amdgcn_wmma_f32_16x16x32_bf16(
                  false, a1, false, b, (short)0, c1[q], false, false);
    }
  }
}

// Store one N-half of dual-M D-layout accumulators into a bf16 LDS tile
// (conversion happens once here, not in the next layer's A loads).
__device__ __forceinline__ void store_half_bf(__bf16* tile, int pitch,
                                              const v8f c0[4], const v8f c1[4],
                                              int nh, int lane, bool act) {
  const int half = lane >> 4, nl = lane & 15;
#pragma unroll
  for (int q = 0; q < 4; ++q) {
    const int n = (nh*4 + q)*16 + nl;
#pragma unroll
    for (int r = 0; r < 8; ++r) {
      float v0 = c0[q][r], v1 = c1[q][r];
      tile[(size_t)(r + 8*half)*pitch + n]      = (__bf16)(act ? silu_f(v0) : v0);
      tile[(size_t)(r + 8*half + 16)*pitch + n] = (__bf16)(act ? silu_f(v1) : v1);
    }
  }
}

// ---------------------------------------------------------------- utility kernels
__global__ void zero_f4(float4* __restrict__ p, int n4) {
  int i = blockIdx.x * 256 + threadIdx.x;
  if (i < n4) p[i] = make_float4(0.f, 0.f, 0.f, 0.f);
}

// Pack f32 row-major W[Kact x 128] (zero-padded to KT*32 rows) into bf16 B
// fragments, kt-major: dword tid = ((kt*8 + nt)*32 + lane)*8 + j.
// lane holds column n = nt*16 + lane%16, K = kt*32 + (j/4)*16 + (lane/16)*8 + 2*(j%4) (+1).
__global__ void pack_w_kernel(const float* __restrict__ W, int Kact, int KT,
                              uint32_t* __restrict__ dst) {
  int tid = blockIdx.x * 256 + threadIdx.x;
  int total = KT * 2048;
  if (tid >= total) return;
  int j    = tid & 7;
  int lane = (tid >> 3) & 31;
  int frag = tid >> 8;
  int nt = frag & 7;
  int kt = frag >> 3;
  int n    = nt*16 + (lane & 15);
  int half = lane >> 4;
  int k0 = kt*32 + (j >> 2)*16 + half*8 + 2*(j & 3);
  int k1 = k0 + 1;
  float f0 = (k0 < Kact) ? W[(size_t)k0*H + n] : 0.0f;
  float f1 = (k1 < Kact) ? W[(size_t)k1*H + n] : 0.0f;
  dst[tid] = pk_bf16(f0, f1);
}

// ---------------------------------------------------------------- edge pipeline
// One wave per 32-edge tile: build the 32x320 bf16 feature tile in LDS once,
// run the 3-layer edge MLP as N-half dual-M WMMA passes, scatter-add messages.
__global__ __launch_bounds__(64) void edge_mlp_scatter(
    const float* __restrict__ x, const int* __restrict__ ei,
    const float* __restrict__ dist,
    const uint32_t* __restrict__ pWm1, const float* __restrict__ bm1,
    const uint32_t* __restrict__ pWm2, const float* __restrict__ bm2,
    const uint32_t* __restrict__ pWm3, const float* __restrict__ bm3,
    float* __restrict__ acc) {
  __shared__ uint32_t sfeat[2][32 * (FP_E/2)];
  __shared__ uint32_t st0[2][32 * (HP/2)];
  __shared__ uint32_t st1[2][32 * (HP/2)];
  const int wave = threadIdx.x >> 5;
  const int lane = threadIdx.x & 31;
  const int half = lane >> 4;
  const int nl   = lane & 15;
  const int e0   = (blockIdx.x * 2 + wave) * 32;

  // ---- build feature tile: rows = 32 edges, cols = [x_row|x_col|radial|pad]
  {
    uint32_t* fd = &sfeat[wave][0];
    const float step  = CUT / (RAD - 1);
    const float coeff = -0.5f / (step * step);
#pragma unroll 4
    for (int m = 0; m < 32; ++m) {
      const int   row = ei[e0 + m];
      const int   col = ei[NE + e0 + m];
      const float d   = dist[e0 + m];
      const float* xr = x + (size_t)row * H;
      const float* xc = x + (size_t)col * H;
      float2 fa = *(const float2*)(xr + 2*lane);
      float2 fb = *(const float2*)(xr + 64 + 2*lane);
      float2 fc = *(const float2*)(xc + 2*lane);
      float2 fe = *(const float2*)(xc + 64 + 2*lane);
      uint32_t* rp = fd + m * (FP_E/2);
      rp[lane]      = pk_bf16(fa.x, fa.y);
      rp[32 + lane] = pk_bf16(fb.x, fb.y);
      rp[64 + lane] = pk_bf16(fc.x, fc.y);
      rp[96 + lane] = pk_bf16(fe.x, fe.y);
      // radial (+ zero pad to 64 slots): rk = 2*lane, 2*lane+1
      int rk0 = 2*lane, rk1 = 2*lane + 1;
      float v0 = 0.f, v1 = 0.f;
      if (rk0 < RAD) { float df = d - rk0*step; v0 = __expf(coeff*df*df); }
      if (rk1 < RAD) { float df = d - rk1*step; v1 = __expf(coeff*df*df); }
      rp[128 + lane] = pk_bf16(v0, v1);
    }
  }
  __syncthreads();

  const __bf16* feat = (const __bf16*)&sfeat[wave][0];
  __bf16* t0 = (__bf16*)&st0[wave][0];
  __bf16* t1 = (__bf16*)&st1[wave][0];
  v8f c0[4], c1[4];

  // Layer 1: silu(feat @ Wm1 + bm1)  (K = 320)
#pragma unroll 1
  for (int nh = 0; nh < 2; ++nh) {
    gemm_half<10>(feat, FP_E, pWm1, bm1, nh, lane, c0, c1);
    store_half_bf(t0, HP, c0, c1, nh, lane, true);
  }
  __syncthreads();

  // Layer 2: silu(h @ Wm2 + bm2)
#pragma unroll 1
  for (int nh = 0; nh < 2; ++nh) {
    gemm_half<4>(t0, HP, pWm2, bm2, nh, lane, c0, c1);
    store_half_bf(t1, HP, c0, c1, nh, lane, true);
  }
  __syncthreads();

  // Layer 3: messages = h @ Wm3 + bm3, scatter-add by col.
  int cols0[8], cols1[8];
#pragma unroll
  for (int r = 0; r < 8; ++r) {
    cols0[r] = ei[NE + e0 + r + 8*half];
    cols1[r] = ei[NE + e0 + 16 + r + 8*half];
  }
#pragma unroll 1
  for (int nh = 0; nh < 2; ++nh) {
    gemm_half<4>(t1, HP, pWm3, bm3, nh, lane, c0, c1);
#pragma unroll
    for (int q = 0; q < 4; ++q) {
      const int n = (nh*4 + q)*16 + nl;
#pragma unroll
      for (int r = 0; r < 8; ++r) {
        unsafeAtomicAdd(acc + (size_t)cols0[r]*H + n, c0[q][r]);
        unsafeAtomicAdd(acc + (size_t)cols1[r]*H + n, c1[q][r]);
      }
    }
  }
}

// ---------------------------------------------------------------- node pipeline
// One wave per block, 32 nodes per wave (single-wave workgroup: barriers = NOP).
__global__ __launch_bounds__(32) void node_mlp(
    const float* __restrict__ x, const float* __restrict__ acc,
    const uint32_t* __restrict__ pWu1, const float* __restrict__ bu1,
    const uint32_t* __restrict__ pWu2, const float* __restrict__ bu2,
    const uint32_t* __restrict__ pWi1, const float* __restrict__ bi1,
    const uint32_t* __restrict__ pWi2, const float* __restrict__ bi2,
    float* __restrict__ out) {
  __shared__ uint32_t sfeat[32 * (FP_N/2)];
  __shared__ uint32_t st0[32 * (HP/2)];
  __shared__ uint32_t st1[32 * (HP/2)];
  const int lane = threadIdx.x & 31;
  const int half = lane >> 4;
  const int nl   = lane & 15;
  const int n0   = blockIdx.x * 32;

  // ---- build feature tile: rows = 32 nodes, cols = [x | aggregated messages]
  {
#pragma unroll 4
    for (int m = 0; m < 32; ++m) {
      const float* xp = x   + (size_t)(n0 + m) * H;
      const float* op = acc + (size_t)(n0 + m) * H;
      float2 fa = *(const float2*)(xp + 2*lane);
      float2 fb = *(const float2*)(xp + 64 + 2*lane);
      float2 fc = *(const float2*)(op + 2*lane);
      float2 fe = *(const float2*)(op + 64 + 2*lane);
      uint32_t* rp = sfeat + m * (FP_N/2);
      rp[lane]      = pk_bf16(fa.x, fa.y);
      rp[32 + lane] = pk_bf16(fb.x, fb.y);
      rp[64 + lane] = pk_bf16(fc.x, fc.y);
      rp[96 + lane] = pk_bf16(fe.x, fe.y);
    }
  }
  __syncthreads();

  const __bf16* feat = (const __bf16*)sfeat;
  __bf16* t0 = (__bf16*)st0;
  __bf16* t1 = (__bf16*)st1;
  v8f c0[4], c1[4];

  // silu([x|out] @ Wu1 + bu1)  (K = 256)
#pragma unroll 1
  for (int nh = 0; nh < 2; ++nh) {
    gemm_half<8>(feat, FP_N, pWu1, bu1, nh, lane, c0, c1);
    store_half_bf(t0, HP, c0, c1, nh, lane, true);
  }
  __syncthreads();

  // u @ Wu2 + bu2  (no act)
#pragma unroll 1
  for (int nh = 0; nh < 2; ++nh) {
    gemm_half<4>(t0, HP, pWu2, bu2, nh, lane, c0, c1);
    store_half_bf(t1, HP, c0, c1, nh, lane, false);
  }
  __syncthreads();

  // silu(u @ Wi1 + bi1)
#pragma unroll 1
  for (int nh = 0; nh < 2; ++nh) {
    gemm_half<4>(t1, HP, pWi1, bi1, nh, lane, c0, c1);
    store_half_bf(t0, HP, c0, c1, nh, lane, true);
  }
  __syncthreads();

  // u @ Wi2 + bi2 -> output
#pragma unroll 1
  for (int nh = 0; nh < 2; ++nh) {
    gemm_half<4>(t0, HP, pWi2, bi2, nh, lane, c0, c1);
#pragma unroll
    for (int q = 0; q < 4; ++q) {
      const int n = (nh*4 + q)*16 + nl;
#pragma unroll
      for (int r = 0; r < 8; ++r) {
        out[(size_t)(n0 + r + 8*half)*H      + n] = c0[q][r];
        out[(size_t)(n0 + 16 + r + 8*half)*H + n] = c1[q][r];
      }
    }
  }
}

// ---------------------------------------------------------------- launcher
extern "C" void kernel_launch(void* const* d_in, const int* in_sizes, int n_in,
                              void* d_out, int out_size, void* d_ws, size_t ws_size,
                              hipStream_t stream) {
  const float* x    = (const float*)d_in[0];
  const int*   ei   = (const int*)d_in[1];
  // d_in[2] = edge_attr (unused by the reference)
  const float* dist = (const float*)d_in[3];
  const float* Wm1  = (const float*)d_in[4];   const float* bm1 = (const float*)d_in[5];
  const float* Wm2  = (const float*)d_in[6];   const float* bm2 = (const float*)d_in[7];
  const float* Wm3  = (const float*)d_in[8];   const float* bm3 = (const float*)d_in[9];
  const float* Wu1  = (const float*)d_in[10];  const float* bu1 = (const float*)d_in[11];
  const float* Wu2  = (const float*)d_in[12];  const float* bu2 = (const float*)d_in[13];
  const float* Wi1  = (const float*)d_in[14];  const float* bi1 = (const float*)d_in[15];
  const float* Wi2  = (const float*)d_in[16];  const float* bi2 = (const float*)d_in[17];
  float* out = (float*)d_out;

  // Workspace: [acc: NN*H f32 = 51.2 MB][packed bf16 weights: ~311 KB]
  float*    accb = (float*)d_ws;
  uint32_t* pw   = (uint32_t*)((char*)d_ws + (size_t)NN * H * sizeof(float));
  uint32_t* pWm1 = pw;                 // KT=10 -> 20480 dwords
  uint32_t* pWm2 = pWm1 + 10 * 2048;   // KT=4  ->  8192
  uint32_t* pWm3 = pWm2 + 4 * 2048;
  uint32_t* pWu1 = pWm3 + 4 * 2048;    // KT=8  -> 16384
  uint32_t* pWu2 = pWu1 + 8 * 2048;
  uint32_t* pWi1 = pWu2 + 4 * 2048;
  uint32_t* pWi2 = pWi1 + 4 * 2048;

  // 1) zero the scatter accumulator (harness does not re-zero between replays)
  {
    int n4 = NN * H / 4;
    hipLaunchKernelGGL(zero_f4, dim3((n4 + 255) / 256), dim3(256), 0, stream,
                       (float4*)accb, n4);
  }

  // 2) pack all weight matrices into WMMA B-fragment layout (bf16, kt-major)
  auto pack1 = [&](const float* W, int Kact, int KT, uint32_t* dst) {
    int total = KT * 2048;
    hipLaunchKernelGGL(pack_w_kernel, dim3((total + 255) / 256), dim3(256), 0,
                       stream, W, Kact, KT, dst);
  };
  pack1(Wm1, 2*H + RAD, 10, pWm1);   // 306 rows zero-padded to 320
  pack1(Wm2, H,         4,  pWm2);
  pack1(Wm3, H,         4,  pWm3);
  pack1(Wu1, 2*H,       8,  pWu1);
  pack1(Wu2, H,         4,  pWu2);
  pack1(Wi1, H,         4,  pWi1);
  pack1(Wi2, H,         4,  pWi2);

  // 3) edge MLP + scatter-add (20000 32-edge wave-tiles, 2 waves/block)
  hipLaunchKernelGGL(edge_mlp_scatter, dim3(NE / 64), dim3(64), 0, stream,
                     x, ei, dist, pWm1, bm1, pWm2, bm2, pWm3, bm3, accb);

  // 4) node MLP (3125 32-node wave-tiles, 1 wave/block)
  hipLaunchKernelGGL(node_mlp, dim3(NN / 32), dim3(32), 0, stream,
                     x, accb, pWu1, bu1, pWu2, bu2, pWi1, bi1, pWi2, bi2, out);
}
